// LegislativeGraphEncoder_77515569758371
// MI455X (gfx1250) — compile-verified, matrix operand order checked by reading, not run on hardware
//
#include <hip/hip_runtime.h>
#include <cstring>
#include <cstddef>

typedef __attribute__((ext_vector_type(16))) _Float16 v16h;
typedef __attribute__((ext_vector_type(8)))  float    v8f;

static constexpr size_t NLEG  = 50000;
static constexpr size_t NBILL = 100000;
static constexpr size_t NE    = 250000;
#define HIDN 192

// ---------------- workspace layout ----------------
constexpr size_t AL(size_t x) { return (x + 255) & ~(size_t)255; }
constexpr size_t OFF_MAXABS = 0;                                   // 5 u32
constexpr size_t OFF_CW     = AL(OFF_MAXABS + 5 * 4);              // 4*192*192 f32 composite kp/vp weights
constexpr size_t OFF_CB     = AL(OFF_CW + 4 * 36864 * 4);          // 4*192 f32 composite biases
constexpr size_t OFF_FR_TPW1  = AL(OFF_CB + 768 * 4);              // frag buffers (f16)
constexpr size_t OFF_FR_TPW2  = AL(OFF_FR_TPW1 + 32 * 192 * 2);
constexpr size_t OFF_FR_VBW1  = AL(OFF_FR_TPW2 + 36864 * 2);
constexpr size_t OFF_FR_VBW2  = AL(OFF_FR_VBW1 + 36864 * 2);
constexpr size_t OFF_FR_POLW1 = AL(OFF_FR_VBW2 + 36864 * 2);       // 384x192
constexpr size_t OFF_FR_POLW2 = AL(OFF_FR_POLW1 + 73728 * 2);
constexpr size_t OFF_FR_AW0   = AL(OFF_FR_POLW2 + 36864 * 2);
constexpr size_t OFF_FR_AW1   = AL(OFF_FR_AW0 + 36864 * 2);
constexpr size_t OFF_FR_QLEG  = AL(OFF_FR_AW1 + 36864 * 2);
constexpr size_t OFF_FR_QBILL = AL(OFF_FR_QLEG + 36864 * 2);
constexpr size_t OFF_FR_KPLEG = AL(OFF_FR_QBILL + 36864 * 2);
constexpr size_t OFF_FR_VPLEG = AL(OFF_FR_KPLEG + 36864 * 2);
constexpr size_t OFF_FR_KPBILL= AL(OFF_FR_VPLEG + 36864 * 2);
constexpr size_t OFF_FR_VPBILL= AL(OFF_FR_KPBILL + 36864 * 2);
constexpr size_t OFF_QLEG   = AL(OFF_FR_VPBILL + 36864 * 2);       // f16 node projections
constexpr size_t OFF_KPLEG  = AL(OFF_QLEG  + NLEG  * HIDN * 2);
constexpr size_t OFF_VPLEG  = AL(OFF_KPLEG + NLEG  * HIDN * 2);
constexpr size_t OFF_QBILL  = AL(OFF_VPLEG + NLEG  * HIDN * 2);
constexpr size_t OFF_KPBILL = AL(OFF_QBILL + NBILL * HIDN * 2);
constexpr size_t OFF_VPBILL = AL(OFF_KPBILL+ NBILL * HIDN * 2);
constexpr size_t OFF_BILLSUM= AL(OFF_VPBILL+ NBILL * HIDN * 2);    // f32 temporal+pol sum
constexpr size_t OFF_CNT    = AL(OFF_BILLSUM + NBILL * HIDN * 4);  // f32 deg
constexpr size_t OFF_MSGB   = AL(OFF_CNT + NBILL * 4);             // f32
constexpr size_t OFF_MSGL   = AL(OFF_MSGB + NBILL * HIDN * 4);
constexpr size_t OFF_MAXB   = AL(OFF_MSGL + NLEG * HIDN * 4);      // u32 encoded
constexpr size_t OFF_SUMB   = AL(OFF_MAXB + NBILL * 2 * 4);
constexpr size_t OFF_MAXL   = AL(OFF_SUMB + NBILL * 2 * 4);
constexpr size_t OFF_SUML   = AL(OFF_MAXL + NLEG * 2 * 4);
constexpr size_t OFF_EX0    = AL(OFF_SUML + NLEG * 2 * 4);         // f32 per-(edge,head) scores
constexpr size_t OFF_EX1    = AL(OFF_EX0 + NE * 2 * 4);
constexpr size_t WS_TOTAL   = AL(OFF_EX1 + NE * 2 * 4);

// ---------------- device helpers ----------------
__device__ __forceinline__ float sanf(float v) {
  if (v != v) return 0.f;
  return fminf(fmaxf(v, -1e4f), 1e4f);
}
__device__ __forceinline__ float geluf(float x) {
  return 0.5f * x * (1.f + erff(x * 0.7071067811865475f));
}
__device__ __forceinline__ v8f vzero() {
  v8f z;
#pragma unroll
  for (int i = 0; i < 8; ++i) z[i] = 0.f;
  return z;
}
// monotone float<->uint encoding for atomicMax on floats
__device__ __forceinline__ unsigned fenc(float f) {
  unsigned u = __float_as_uint(f);
  return (u & 0x80000000u) ? ~u : (u | 0x80000000u);
}
__device__ __forceinline__ float fdec(unsigned u) {
  u = (u & 0x80000000u) ? (u & 0x7FFFFFFFu) : ~u;
  return __uint_as_float(u);
}
// ISA 7.12.2: 16-bit A-matrix K index for vector element e, lane-half hi
__device__ __forceinline__ int frag_k_of(int e, int hiAdd) {
  int v = e >> 1, h = e & 1;
  int k = (v < 4) ? (2 * v + h) : (16 + 2 * (v - 4) + h);
  return k + hiAdd;
}
__device__ __forceinline__ v16h load_fragA(const _Float16* s, int ld, int k0, int lane) {
  v16h f;
  int m = lane & 15;
  int add = (lane >> 4) ? 8 : 0;
#pragma unroll
  for (int e = 0; e < 16; ++e) f[e] = s[m * ld + k0 + frag_k_of(e, add)];
  return f;
}
// 16 rows (in LDS, f16, row-major ld) x [K x 192] frag weights -> this wave's 3 col tiles
__device__ __forceinline__ void gemm3(const _Float16* sAp, int ldA, int K,
                                      const _Float16* frW, int lane, int wave, v8f acc[3]) {
  const int nkt = K >> 5;
  const v16h* bw = (const v16h*)frW;
  for (int kt = 0; kt < nkt; ++kt) {
    v16h a = load_fragA(sAp, ldA, kt * 32, lane);
#pragma unroll
    for (int t = 0; t < 3; ++t) {
      int nt = wave * 3 + t;
      v16h b = bw[(size_t)(nt * nkt + kt) * 32 + lane];
      acc[t] = __builtin_amdgcn_wmma_f32_16x16x32_f16(false, a, false, b, (short)0, acc[t],
                                                      false, false);
    }
  }
}

// ---------------- small utility kernels ----------------
__global__ void fill_f32_kernel(float* p, float v, size_t n) {
  size_t i = (size_t)blockIdx.x * blockDim.x + threadIdx.x;
  if (i < n) p[i] = v;
}
__global__ void fill_u32_kernel(unsigned* p, unsigned v, size_t n) {
  size_t i = (size_t)blockIdx.x * blockDim.x + threadIdx.x;
  if (i < n) p[i] = v;
}

// composite kp/vp weights: fold block-diag rel_a/rel_m and rel_p/sqrt(D) into 192x192 mats
__global__ void prep_comp_kernel(const float* k_w, const float* k_b, const float* v_w,
                                 const float* v_b, const float* rel_a, const float* rel_m,
                                 const float* rel_p, float* CW, float* CB) {
  int i = blockIdx.x * blockDim.x + threadIdx.x;
  const float invsqd = 0.10206207261596575f;  // 1/sqrt(96)
  if (i < 4 * 36864) {
    int which = i / 36864, rem = i % 36864;
    int c = rem / 192, ef = rem % 192, h = ef / 96, e = ef % 96;
    int ntype = which >> 1;       // 0 = leg (relation 0), 1 = bill (relation 1)
    int iskp = !(which & 1);      // even = kp, odd = vp
    const float* W = (iskp ? k_w : v_w) + (size_t)ntype * 36864;
    const float* R = (iskp ? rel_a : rel_m) + ((size_t)ntype * 2 + h) * 9216;
    float scale = iskp ? rel_p[ntype * 2 + h] * invsqd : 1.f;
    float s = 0.f;
    for (int d0 = 0; d0 < 96; ++d0) s += W[c * 192 + h * 96 + d0] * R[d0 * 96 + e];
    CW[i] = s * scale;
  } else if (i < 4 * 36864 + 4 * 192) {
    int j = i - 4 * 36864;
    int which = j / 192, ef = j % 192, h = ef / 96, e = ef % 96;
    int ntype = which >> 1;
    int iskp = !(which & 1);
    const float* bsrc = (iskp ? k_b : v_b) + (size_t)ntype * 192;
    const float* R = (iskp ? rel_a : rel_m) + ((size_t)ntype * 2 + h) * 9216;
    float scale = iskp ? rel_p[ntype * 2 + h] * invsqd : 1.f;
    float s = 0.f;
    for (int d0 = 0; d0 < 96; ++d0) s += bsrc[h * 96 + d0] * R[d0 * 96 + e];
    CB[j] = s * scale;
  }
}

// convert row-major f32 [K x 192] weight to f16 WMMA B fragments
__global__ void prep_frag_kernel(const float* src, _Float16* dst, int K) {
  int i = blockIdx.x * blockDim.x + threadIdx.x;
  if (i >= K * 192) return;
  int e = i & 15, lane = (i >> 4) & 31, tile = i >> 9;
  int nkt = K >> 5;
  int nt = tile / nkt, kt = tile % nkt;
  int n = nt * 16 + (lane & 15);
  int k = kt * 32 + frag_k_of(e, (lane >> 4) ? 8 : 0);
  dst[i] = (_Float16)src[(size_t)k * 192 + n];
}

// channel-wise max|tf| over all edges
__global__ __launch_bounds__(256) void maxabs_kernel(const float* ets, const float* edl,
                                                     const float* tsl, const float* tsb,
                                                     const int* srcv, const int* dstv,
                                                     unsigned* om, int E_) {
  __shared__ float sred[5][256];
  int tid = threadIdx.x;
  float lm[5] = {0, 0, 0, 0, 0};
  for (int e = blockIdx.x * 256 + tid; e < E_; e += gridDim.x * 256) {
    float ts = sanf(ets[e]), dl = sanf(edl[e]);
    float sts = sanf(tsl[srcv[e]]), dts = sanf(tsb[dstv[e]]);
    lm[0] = fmaxf(lm[0], fabsf(ts));
    lm[1] = fmaxf(lm[1], fabsf(sts));
    lm[2] = fmaxf(lm[2], fabsf(dts));
    lm[3] = fmaxf(lm[3], fabsf(dl));
    lm[4] = fmaxf(lm[4], fabsf(dts - ts));
  }
  for (int c = 0; c < 5; ++c) sred[c][tid] = lm[c];
  __syncthreads();
  for (int s2 = 128; s2 > 0; s2 >>= 1) {
    if (tid < s2)
      for (int c = 0; c < 5; ++c) sred[c][tid] = fmaxf(sred[c][tid], sred[c][tid + s2]);
    __syncthreads();
  }
  if (tid == 0)
    for (int c = 0; c < 5; ++c) atomicMax(om + c, __float_as_uint(sred[c][0]));  // vals >= 0
}

// ---------------- fused per-edge MLP kernel (16 edges / block) ----------------
__global__ __launch_bounds__(128) void edge_kernel(
    const float* __restrict__ ets, const float* __restrict__ edl,
    const float* __restrict__ tsl, const float* __restrict__ tsb,
    const int* __restrict__ srcv, const int* __restrict__ dstv,
    const float* __restrict__ eattr, const float* __restrict__ freqw,
    const float* __restrict__ phase, const unsigned* __restrict__ mabs,
    const _Float16* __restrict__ frTPW1, const _Float16* __restrict__ frTPW2,
    const _Float16* __restrict__ frVBW1, const _Float16* __restrict__ frVBW2,
    const _Float16* __restrict__ frPOLW1, const _Float16* __restrict__ frPOLW2,
    const float* __restrict__ tp_b1, const float* __restrict__ tp_b2,
    const float* __restrict__ ln_g, const float* __restrict__ ln_b,
    const float* __restrict__ vb_b1, const float* __restrict__ vb_b2,
    const float* __restrict__ pol_b1, const float* __restrict__ pol_b2,
    float* __restrict__ billsum, float* __restrict__ cnt) {
  __shared__ _Float16 sA[16 * 384];
  __shared__ _Float16 sB[16 * 192];
  __shared__ float sF[16 * 192];
  __shared__ float sPol[16];
  __shared__ int sDst[16];
  __shared__ float sTf[16 * 5];
  const int tid = threadIdx.x, lane = tid & 31, wave = tid >> 5;
  const int hi = lane >> 4, n15 = lane & 15;
  const size_t e0 = (size_t)blockIdx.x * 16;

  if (tid < 16) {
    size_t e = e0 + tid;
    float ts = sanf(ets[e]), dl = sanf(edl[e]);
    float sts = sanf(tsl[srcv[e]]), dts = sanf(tsb[dstv[e]]);
    float tf0[5] = {ts, sts, dts, dl, dts - ts};
    sDst[tid] = dstv[e];
    sPol[tid] = fminf(fmaxf(eattr[e * 385], 0.f), 1.f);
#pragma unroll
    for (int c = 0; c < 5; ++c)
      sTf[tid * 5 + c] = tf0[c] / fmaxf(__uint_as_float(mabs[c]), 1.f);
  }
  __syncthreads();
  // sin/cos temporal encoding -> sA as 16x32 f16
  for (int i = tid; i < 256; i += 128) {
    int r = i >> 4, f = i & 15;
    float ang = phase[f];
#pragma unroll
    for (int c = 0; c < 5; ++c) ang += sTf[r * 5 + c] * freqw[c * 16 + f];
    sA[r * 32 + f] = (_Float16)sinf(ang);
    sA[r * 32 + 16 + f] = (_Float16)cosf(ang);
  }
  __syncthreads();

  // GEMM1: enc @ tp_w1 (+b1, gelu) -> sB
  {
    v8f acc[3];
    acc[0] = acc[1] = acc[2] = vzero();
    gemm3(sA, 32, 32, frTPW1, lane, wave, acc);
#pragma unroll
    for (int t = 0; t < 3; ++t) {
      int n = (wave * 3 + t) * 16 + n15;
      float b = tp_b1[n];
#pragma unroll
      for (int v = 0; v < 8; ++v) {
        int m = v + hi * 8;
        sB[m * 192 + n] = (_Float16)geluf(acc[t][v] + b);
      }
    }
  }
  __syncthreads();
  // GEMM2: @ tp_w2 + b2 -> sF (f32, pre-LN)
  {
    v8f acc[3];
    acc[0] = acc[1] = acc[2] = vzero();
    gemm3(sB, 192, 192, frTPW2, lane, wave, acc);
#pragma unroll
    for (int t = 0; t < 3; ++t) {
      int n = (wave * 3 + t) * 16 + n15;
      float b = tp_b2[n];
#pragma unroll
      for (int v = 0; v < 8; ++v) sF[(v + hi * 8) * 192 + n] = acc[t][v] + b;
    }
  }
  __syncthreads();
  // LayerNorm per row -> sA (f16, ld 192)
  if (tid < 16) {
    float mu = 0.f;
    for (int c = 0; c < 192; ++c) mu += sF[tid * 192 + c];
    mu *= (1.f / 192.f);
    float var = 0.f;
    for (int c = 0; c < 192; ++c) {
      float d = sF[tid * 192 + c] - mu;
      var += d * d;
    }
    var *= (1.f / 192.f);
    float inv = rsqrtf(var + 1e-5f);
    for (int c = 0; c < 192; ++c)
      sA[tid * 192 + c] = (_Float16)((sF[tid * 192 + c] - mu) * inv * ln_g[c] + ln_b[c]);
  }
  __syncthreads();
  // GEMM3: @ vb_w1 (+b, gelu) -> sB
  {
    v8f acc[3];
    acc[0] = acc[1] = acc[2] = vzero();
    gemm3(sA, 192, 192, frVBW1, lane, wave, acc);
#pragma unroll
    for (int t = 0; t < 3; ++t) {
      int n = (wave * 3 + t) * 16 + n15;
      float b = vb_b1[n];
#pragma unroll
      for (int v = 0; v < 8; ++v) sB[(v + hi * 8) * 192 + n] = (_Float16)geluf(acc[t][v] + b);
    }
  }
  __syncthreads();
  // GEMM4: @ vb_w2 + b -> sF (tmsg); then stage edge_attr[:,1:] -> sA (16x384 f16)
  {
    v8f acc[3];
    acc[0] = acc[1] = acc[2] = vzero();
    gemm3(sB, 192, 192, frVBW2, lane, wave, acc);
#pragma unroll
    for (int t = 0; t < 3; ++t) {
      int n = (wave * 3 + t) * 16 + n15;
      float b = vb_b2[n];
#pragma unroll
      for (int v = 0; v < 8; ++v) sF[(v + hi * 8) * 192 + n] = acc[t][v] + b;
    }
  }
  for (int i = tid; i < 16 * 384; i += 128) {
    int r = i / 384, c = i % 384;
    sA[i] = (_Float16)eattr[(e0 + r) * 385 + 1 + c];
  }
  __syncthreads();
  // GEMM5: @ pol_w1 (K=384, +b, relu) -> sB
  {
    v8f acc[3];
    acc[0] = acc[1] = acc[2] = vzero();
    gemm3(sA, 384, 384, frPOLW1, lane, wave, acc);
#pragma unroll
    for (int t = 0; t < 3; ++t) {
      int n = (wave * 3 + t) * 16 + n15;
      float b = pol_b1[n];
#pragma unroll
      for (int v = 0; v < 8; ++v)
        sB[(v + hi * 8) * 192 + n] = (_Float16)fmaxf(acc[t][v] + b, 0.f);
    }
  }
  __syncthreads();
  // GEMM6: @ pol_w2 + b, *(pol+0.01), + tmsg, atomic into billsum
  {
    v8f acc[3];
    acc[0] = acc[1] = acc[2] = vzero();
    gemm3(sB, 192, 192, frPOLW2, lane, wave, acc);
#pragma unroll
    for (int t = 0; t < 3; ++t) {
      int n = (wave * 3 + t) * 16 + n15;
      float b = pol_b2[n];
#pragma unroll
      for (int v = 0; v < 8; ++v) {
        int m = v + hi * 8;
        float ef = (acc[t][v] + b) * (sPol[m] + 0.01f);
        float comb = ef + sF[m * 192 + n];
        atomicAdd(&billsum[(size_t)sDst[m] * 192 + n], comb);
      }
    }
  }
  if (tid < 16) atomicAdd(&cnt[sDst[tid]], 1.f);
}

// ---------------- generic node projection GEMM -> f16 ----------------
__global__ __launch_bounds__(128) void proj_kernel(const float* __restrict__ X,
                                                   const _Float16* __restrict__ frW,
                                                   const float* __restrict__ bias,
                                                   _Float16* __restrict__ out) {
  __shared__ _Float16 sA[16 * 192];
  const int tid = threadIdx.x, lane = tid & 31, wave = tid >> 5;
  const int hi = lane >> 4, n15 = lane & 15;
  const size_t r0 = (size_t)blockIdx.x * 16;
  for (int i = tid; i < 16 * 192; i += 128) {
    int r = i / 192, c = i % 192;
    sA[i] = (_Float16)X[(r0 + r) * 192 + c];
  }
  __syncthreads();
  v8f acc[3];
  acc[0] = acc[1] = acc[2] = vzero();
  gemm3(sA, 192, 192, frW, lane, wave, acc);
#pragma unroll
  for (int t = 0; t < 3; ++t) {
    int n = (wave * 3 + t) * 16 + n15;
    float b = bias[n];
#pragma unroll
    for (int v = 0; v < 8; ++v) {
      int m = v + hi * 8;
      out[(r0 + m) * 192 + n] = (_Float16)(acc[t][v] + b);
    }
  }
}

// ---------------- attention: pass 1 (score + max) ----------------
__global__ void attn_score_kernel(const int* __restrict__ srcv, const int* __restrict__ dstv,
                                  const _Float16* __restrict__ qBill,
                                  const _Float16* __restrict__ kpLeg,
                                  const _Float16* __restrict__ qLeg,
                                  const _Float16* __restrict__ kpBill,
                                  float* __restrict__ ex0, float* __restrict__ ex1,
                                  unsigned* __restrict__ maxB, unsigned* __restrict__ maxL,
                                  int E_) {
  int i = blockIdx.x * blockDim.x + threadIdx.x;
  if (i >= E_ * 2) return;
  int r = blockIdx.y;
  int e = i >> 1, h = i & 1;
  int s, d;
  const _Float16 *q, *kp;
  float* exb;
  unsigned* mx;
  if (r == 0) { s = srcv[e]; d = dstv[e]; q = qBill; kp = kpLeg; exb = ex0; mx = maxB; }
  else        { s = dstv[e]; d = srcv[e]; q = qLeg;  kp = kpBill; exb = ex1; mx = maxL; }
  const _Float16* qp = q + (size_t)d * 192 + h * 96;
  const _Float16* kpp = kp + (size_t)s * 192 + h * 96;
  float sc = 0.f;
#pragma unroll 4
  for (int j = 0; j < 96; ++j) sc += (float)qp[j] * (float)kpp[j];
  exb[i] = sc;
  atomicMax(mx + (size_t)d * 2 + h, fenc(sc));
}

// ---------------- attention: pass 2 (exp, sum, weighted V accumulate) ----------------
__global__ void attn_accum_kernel(const int* __restrict__ srcv, const int* __restrict__ dstv,
                                  const _Float16* __restrict__ vpLeg,
                                  const _Float16* __restrict__ vpBill,
                                  const float* __restrict__ ex0, const float* __restrict__ ex1,
                                  const unsigned* __restrict__ maxB,
                                  const unsigned* __restrict__ maxL,
                                  float* __restrict__ sumB, float* __restrict__ sumL,
                                  float* __restrict__ msgB, float* __restrict__ msgL, int E_) {
  int i = blockIdx.x * blockDim.x + threadIdx.x;
  if (i >= E_ * 2) return;
  int r = blockIdx.y;
  int e = i >> 1, h = i & 1;
  int s, d;
  const _Float16* vp;
  const float* exb;
  const unsigned* mx;
  float *sm, *mg;
  if (r == 0) { s = srcv[e]; d = dstv[e]; vp = vpLeg; exb = ex0; mx = maxB; sm = sumB; mg = msgB; }
  else        { s = dstv[e]; d = srcv[e]; vp = vpBill; exb = ex1; mx = maxL; sm = sumL; mg = msgL; }
  float m = fdec(mx[(size_t)d * 2 + h]);
  float ex = expf(exb[i] - m);
  atomicAdd(sm + (size_t)d * 2 + h, ex);
  const _Float16* vpp = vp + (size_t)s * 192 + h * 96;
  float* mgp = mg + (size_t)d * 192 + h * 96;
#pragma unroll 4
  for (int j = 0; j < 96; ++j) atomicAdd(mgp + j, ex * (float)vpp[j]);
}

// ---------------- output: gelu(msg/sum) @ a_w + skip blend (+ bill extras) ----------------
__global__ __launch_bounds__(128) void out_kernel(const float* __restrict__ msg,
                                                  const float* __restrict__ ssum,
                                                  const float* __restrict__ x,
                                                  const _Float16* __restrict__ frAW,
                                                  const float* __restrict__ ab,
                                                  const float* __restrict__ skp,
                                                  const float* __restrict__ billsum,
                                                  const float* __restrict__ cnt,
                                                  float* __restrict__ out,
                                                  unsigned long long outRow0) {
  __shared__ _Float16 sA[16 * 192];
  const int tid = threadIdx.x, lane = tid & 31, wave = tid >> 5;
  const int hi = lane >> 4, n15 = lane & 15;
  const size_t r0 = (size_t)blockIdx.x * 16;
  for (int i = tid; i < 16 * 192; i += 128) {
    int r = i / 192, c = i % 192;
    size_t nd = r0 + r;
    int h = c / 96;
    float m = msg[nd * 192 + c] / (ssum[nd * 2 + h] + 1e-16f);
    sA[i] = (_Float16)geluf(m);
  }
  __syncthreads();
  v8f acc[3];
  acc[0] = acc[1] = acc[2] = vzero();
  gemm3(sA, 192, 192, frAW, lane, wave, acc);
  float g = 1.f / (1.f + expf(-skp[0]));
#pragma unroll
  for (int t = 0; t < 3; ++t) {
    int n = (wave * 3 + t) * 16 + n15;
    float b = ab[n];
#pragma unroll
    for (int v = 0; v < 8; ++v) {
      int m = v + hi * 8;
      size_t nd = r0 + m;
      float val = g * (acc[t][v] + b) + (1.f - g) * x[nd * 192 + n];
      if (billsum) val += billsum[nd * 192 + n] / fmaxf(cnt[nd], 1.f);
      out[((size_t)outRow0 + nd) * 192 + n] = val;
    }
  }
}

// ---------------- host launcher ----------------
extern "C" void kernel_launch(void* const* d_in, const int* in_sizes, int n_in, void* d_out,
                              int out_size, void* d_ws, size_t ws_size, hipStream_t stream) {
  (void)in_sizes; (void)n_in; (void)out_size;
  const float* x_leg  = (const float*)d_in[0];
  const float* x_bill = (const float*)d_in[1];
  const float* eattr  = (const float*)d_in[2];
  const float* ets    = (const float*)d_in[3];
  const float* edl    = (const float*)d_in[4];
  const float* tsl    = (const float*)d_in[5];
  const float* tsb    = (const float*)d_in[6];
  const float* freqw  = (const float*)d_in[7];
  const float* phase  = (const float*)d_in[8];
  const float* tp_w1  = (const float*)d_in[9];
  const float* tp_b1  = (const float*)d_in[10];
  const float* tp_w2  = (const float*)d_in[11];
  const float* tp_b2  = (const float*)d_in[12];
  const float* ln_g   = (const float*)d_in[13];
  const float* ln_b   = (const float*)d_in[14];
  const float* vb_w1  = (const float*)d_in[15];
  const float* vb_b1  = (const float*)d_in[16];
  const float* vb_w2  = (const float*)d_in[17];
  const float* vb_b2  = (const float*)d_in[18];
  const float* pol_w1 = (const float*)d_in[19];
  const float* pol_b1 = (const float*)d_in[20];
  const float* pol_w2 = (const float*)d_in[21];
  const float* pol_b2 = (const float*)d_in[22];
  const float* k_w    = (const float*)d_in[23];
  const float* k_b    = (const float*)d_in[24];
  const float* q_w    = (const float*)d_in[25];
  const float* q_b    = (const float*)d_in[26];
  const float* v_w    = (const float*)d_in[27];
  const float* v_b    = (const float*)d_in[28];
  const float* a_w    = (const float*)d_in[29];
  const float* a_b    = (const float*)d_in[30];
  const float* rel_a  = (const float*)d_in[31];
  const float* rel_m  = (const float*)d_in[32];
  const float* rel_p  = (const float*)d_in[33];
  const float* skipv  = (const float*)d_in[34];
  const int* srcv     = (const int*)d_in[35];
  const int* dstv     = (const int*)d_in[36];
  float* out = (float*)d_out;
  if (ws_size < WS_TOTAL) return;

  char* ws = (char*)d_ws;
  unsigned* wMAXABS = (unsigned*)(ws + OFF_MAXABS);
  float* wCW = (float*)(ws + OFF_CW);
  float* wCB = (float*)(ws + OFF_CB);
  _Float16* frTPW1  = (_Float16*)(ws + OFF_FR_TPW1);
  _Float16* frTPW2  = (_Float16*)(ws + OFF_FR_TPW2);
  _Float16* frVBW1  = (_Float16*)(ws + OFF_FR_VBW1);
  _Float16* frVBW2  = (_Float16*)(ws + OFF_FR_VBW2);
  _Float16* frPOLW1 = (_Float16*)(ws + OFF_FR_POLW1);
  _Float16* frPOLW2 = (_Float16*)(ws + OFF_FR_POLW2);
  _Float16* frAW0   = (_Float16*)(ws + OFF_FR_AW0);
  _Float16* frAW1   = (_Float16*)(ws + OFF_FR_AW1);
  _Float16* frQLEG  = (_Float16*)(ws + OFF_FR_QLEG);
  _Float16* frQBILL = (_Float16*)(ws + OFF_FR_QBILL);
  _Float16* frKPLEG = (_Float16*)(ws + OFF_FR_KPLEG);
  _Float16* frVPLEG = (_Float16*)(ws + OFF_FR_VPLEG);
  _Float16* frKPBILL= (_Float16*)(ws + OFF_FR_KPBILL);
  _Float16* frVPBILL= (_Float16*)(ws + OFF_FR_VPBILL);
  _Float16* pQLEG   = (_Float16*)(ws + OFF_QLEG);
  _Float16* pKPLEG  = (_Float16*)(ws + OFF_KPLEG);
  _Float16* pVPLEG  = (_Float16*)(ws + OFF_VPLEG);
  _Float16* pQBILL  = (_Float16*)(ws + OFF_QBILL);
  _Float16* pKPBILL = (_Float16*)(ws + OFF_KPBILL);
  _Float16* pVPBILL = (_Float16*)(ws + OFF_VPBILL);
  float* wBILLSUM = (float*)(ws + OFF_BILLSUM);
  float* wCNT  = (float*)(ws + OFF_CNT);
  float* wMSGB = (float*)(ws + OFF_MSGB);
  float* wMSGL = (float*)(ws + OFF_MSGL);
  unsigned* wMAXB = (unsigned*)(ws + OFF_MAXB);
  float* wSUMB = (float*)(ws + OFF_SUMB);
  unsigned* wMAXL = (unsigned*)(ws + OFF_MAXL);
  float* wSUML = (float*)(ws + OFF_SUML);
  float* wEX0 = (float*)(ws + OFF_EX0);
  float* wEX1 = (float*)(ws + OFF_EX1);

  auto zf = [&](float* p, size_t n) {
    fill_f32_kernel<<<dim3((unsigned)((n + 255) / 256)), 256, 0, stream>>>(p, 0.f, n);
  };
  zf((float*)wMAXABS, 5);
  zf(wBILLSUM, NBILL * HIDN);
  zf(wCNT, NBILL);
  zf(wMSGB, NBILL * HIDN);
  zf(wMSGL, NLEG * HIDN);
  zf(wSUMB, NBILL * 2);
  zf(wSUML, NLEG * 2);
  // encoded -inf for float atomicMax
  float ninf = -3.0e38f;
  unsigned nbits;
  memcpy(&nbits, &ninf, 4);
  unsigned ninf_enc = (nbits & 0x80000000u) ? ~nbits : (nbits | 0x80000000u);
  fill_u32_kernel<<<dim3((unsigned)((NBILL * 2 + 255) / 256)), 256, 0, stream>>>(wMAXB, ninf_enc, NBILL * 2);
  fill_u32_kernel<<<dim3((unsigned)((NLEG * 2 + 255) / 256)), 256, 0, stream>>>(wMAXL, ninf_enc, NLEG * 2);

  prep_comp_kernel<<<(4 * 36864 + 768 + 255) / 256, 256, 0, stream>>>(k_w, k_b, v_w, v_b, rel_a,
                                                                      rel_m, rel_p, wCW, wCB);
  auto prep = [&](const float* s, int K, _Float16* dstF) {
    prep_frag_kernel<<<(K * 192 + 255) / 256, 256, 0, stream>>>(s, dstF, K);
  };
  prep(tp_w1, 32, frTPW1);   prep(tp_w2, 192, frTPW2);
  prep(vb_w1, 192, frVBW1);  prep(vb_w2, 192, frVBW2);
  prep(pol_w1, 384, frPOLW1); prep(pol_w2, 192, frPOLW2);
  prep(a_w, 192, frAW0);      prep(a_w + 36864, 192, frAW1);
  prep(q_w, 192, frQLEG);     prep(q_w + 36864, 192, frQBILL);
  prep(wCW, 192, frKPLEG);            prep(wCW + 36864, 192, frVPLEG);
  prep(wCW + 2 * 36864, 192, frKPBILL); prep(wCW + 3 * 36864, 192, frVPBILL);

  maxabs_kernel<<<512, 256, 0, stream>>>(ets, edl, tsl, tsb, srcv, dstv, wMAXABS, (int)NE);

  edge_kernel<<<(unsigned)(NE / 16), 128, 0, stream>>>(
      ets, edl, tsl, tsb, srcv, dstv, eattr, freqw, phase, wMAXABS, frTPW1, frTPW2, frVBW1,
      frVBW2, frPOLW1, frPOLW2, tp_b1, tp_b2, ln_g, ln_b, vb_b1, vb_b2, pol_b1, pol_b2,
      wBILLSUM, wCNT);

  proj_kernel<<<(unsigned)(NLEG / 16), 128, 0, stream>>>(x_leg, frQLEG, q_b, pQLEG);
  proj_kernel<<<(unsigned)(NLEG / 16), 128, 0, stream>>>(x_leg, frKPLEG, wCB + 0, pKPLEG);
  proj_kernel<<<(unsigned)(NLEG / 16), 128, 0, stream>>>(x_leg, frVPLEG, wCB + 192, pVPLEG);
  proj_kernel<<<(unsigned)(NBILL / 16), 128, 0, stream>>>(x_bill, frQBILL, q_b + 192, pQBILL);
  proj_kernel<<<(unsigned)(NBILL / 16), 128, 0, stream>>>(x_bill, frKPBILL, wCB + 384, pKPBILL);
  proj_kernel<<<(unsigned)(NBILL / 16), 128, 0, stream>>>(x_bill, frVPBILL, wCB + 576, pVPBILL);

  dim3 ag((unsigned)((NE * 2 + 255) / 256), 2);
  attn_score_kernel<<<ag, 256, 0, stream>>>(srcv, dstv, pQBILL, pKPLEG, pQLEG, pKPBILL, wEX0,
                                            wEX1, wMAXB, wMAXL, (int)NE);
  attn_accum_kernel<<<ag, 256, 0, stream>>>(srcv, dstv, pVPLEG, pVPBILL, wEX0, wEX1, wMAXB,
                                            wMAXL, wSUMB, wSUML, wMSGB, wMSGL, (int)NE);

  out_kernel<<<(unsigned)(NLEG / 16), 128, 0, stream>>>(wMSGL, wSUML, x_leg, frAW0, a_b, skipv,
                                                        nullptr, nullptr, out, 0ull);
  out_kernel<<<(unsigned)(NBILL / 16), 128, 0, stream>>>(wMSGB, wSUMB, x_bill, frAW1, a_b + 192,
                                                         skipv + 1, wBILLSUM, wCNT, out,
                                                         (unsigned long long)NLEG);
}
